// CustomMultiHeadAttentionStoich_63539746177258
// MI455X (gfx1250) — matrix operand order repaired and checked by reading, not compiled, not run on hardware
//
#include <hip/hip_runtime.h>
#include <hip/hip_bf16.h>

// ---------------------------------------------------------------------------
// Types for WMMA fragments (gfx1250, wave32)
// ---------------------------------------------------------------------------
typedef __attribute__((ext_vector_type(16))) _Float16 v16h;
typedef __attribute__((ext_vector_type(8)))  _Float16 h8;
typedef __attribute__((ext_vector_type(8)))  float    v8f;
typedef __attribute__((ext_vector_type(4)))  int      v4i;

// Builtin signature (probe-confirmed by round-2 diagnostic):
//   __builtin_amdgcn_global_load_async_to_lds_b128(v4i AS1*, v4i AS3*, Ii, Ii)
typedef __attribute__((address_space(1))) v4i as1_v4i;
typedef __attribute__((address_space(3))) v4i as3_v4i;

#define D_MODEL 1024
#define NHEAD   16
#define HEAD_DIM 64

// ---- gfx1250 async global->LDS copy (guarded; fallback = sync copy) -------
#if defined(__has_builtin)
#  if __has_builtin(__builtin_amdgcn_global_load_async_to_lds_b128)
#    define HAVE_ASYNC_LDS 1
#  endif
#  if __has_builtin(__builtin_amdgcn_s_wait_asynccnt)
#    define HAVE_WAIT_ASYNC 1
#  endif
#endif

__device__ inline void cp16_async_or_sync(const _Float16* g, _Float16* l) {
#if defined(HAVE_ASYNC_LDS)
    __builtin_amdgcn_global_load_async_to_lds_b128(
        (as1_v4i*)g, (as3_v4i*)l, 0, 0);
#else
    *(h8*)l = *(const h8*)g;
#endif
}

// Wait until at most N async ops remain outstanding (immediate).
#if defined(HAVE_ASYNC_LDS)
#  if defined(HAVE_WAIT_ASYNC)
#    define WAIT_ASYNC(n) __builtin_amdgcn_s_wait_asynccnt(n)
#  else
#    define WAIT_ASYNC(n) asm volatile("s_wait_asynccnt %0" :: "i"(n) : "memory")
#  endif
#else
#  define WAIT_ASYNC(n) ((void)0)
#endif

__device__ inline v8f wmma_f16(v16h a, v16h b, v8f c) {
    // D = A(16x32 f16) * B(32x16 f16) + C(16x16 f32)
    return __builtin_amdgcn_wmma_f32_16x16x32_f16(
        /*neg_a=*/false, a, /*neg_b=*/false, b,
        /*c_mod=*/(short)0, c, /*reuse_a=*/false, /*reuse_b=*/false);
}

union AFrag { v16h v; h8 h[2]; };

// A fragment: 16x32 f16, LDS stored row-major [m][k] with stride ld (halves).
// Per ISA 7.12.2: lane l holds row m=l%16; lanes<16 get k {0..7,16..23},
// lanes>=16 get k {8..15,24..31}. Two contiguous 16B DS loads per lane.
__device__ inline v16h load_frag_a(const _Float16* lds, int ld, int mBase,
                                   int kBase, int lane) {
    const int m  = mBase + (lane & 15);
    const int hi = (lane >> 4) & 1;
    const _Float16* p = lds + m * ld + kBase + hi * 8;
    AFrag f;
    f.h[0] = *(const h8*)(p);
    f.h[1] = *(const h8*)(p + 16);
    return f.v;
}

// B fragment: 32x16 f16 (KxN), LDS stored as [n][k] with stride ld (halves).
// Lane l holds column n=l%16; lanes<16 k {0..7,16..23}, lanes>=16 k {8..15,24..31}.
__device__ inline v16h load_frag_b(const _Float16* lds, int ld, int nBase,
                                   int kBase, int lane) {
    const int n  = nBase + (lane & 15);
    const int hi = (lane >> 4) & 1;
    const _Float16* p = lds + n * ld + kBase + hi * 8;
    AFrag f;
    f.h[0] = *(const h8*)(p);
    f.h[1] = *(const h8*)(p + 16);
    return f.v;
}

// ---------------------------------------------------------------------------
// f32 -> f16 conversion (one-time pass over inputs and weights)
// ---------------------------------------------------------------------------
__global__ __launch_bounds__(256) void cvt_f32_f16_kernel(
    const float* __restrict__ src, _Float16* __restrict__ dst, int n)
{
    const int i = (blockIdx.x * 256 + threadIdx.x) * 8;
    if (i >= n) return;
    float4 a = ((const float4*)(src + i))[0];
    float4 b = ((const float4*)(src + i))[1];
    union { h8 h; _Float16 e[8]; } u;
    u.e[0] = (_Float16)a.x; u.e[1] = (_Float16)a.y;
    u.e[2] = (_Float16)a.z; u.e[3] = (_Float16)a.w;
    u.e[4] = (_Float16)b.x; u.e[5] = (_Float16)b.y;
    u.e[6] = (_Float16)b.z; u.e[7] = (_Float16)b.w;
    *(h8*)(dst + i) = u.h;
}

// ---------------------------------------------------------------------------
// All-f16 GEMM: Y = A @ Bw^T + bias ; A f16 [M x K], Bw f16 [N x K].
// QKV=true : write f16 output in attention layout [B][H][T][HEAD_DIM]
// QKV=false: write f32 output [M x N]
// Block tile 128x128, BK=64, 256 threads = 8 wave32.
// Double-buffered LDS: async tile k+1 copies overlap tile k WMMA.
// ---------------------------------------------------------------------------
#define BM 128
#define BN 128
#define BK 64
#define LDK 72   // 64 + 8 pad halves; 144B row stride, 16B aligned

template <bool QKV>
__global__ __launch_bounds__(256) void gemm_f16_kernel(
    const _Float16* __restrict__ A, const _Float16* __restrict__ Bw,
    const float* __restrict__ bias, void* __restrict__ OutP,
    int M, int N, int K, int T)
{
    __shared__ _Float16 As[2][BM * LDK];   // 2 x 18 KB
    __shared__ _Float16 Bs[2][BN * LDK];   // 2 x 18 KB

    const int tid  = threadIdx.x;
    const int lane = tid & 31;
    const int wave = tid >> 5;       // 0..7
    const int wm   = wave & 1;       // 2 m-waves * 64 rows
    const int wn   = wave >> 1;      // 4 n-waves * 32 cols
    const int mBlk = blockIdx.x * BM;
    const int nBlk = blockIdx.y * BN;

    v8f zero = {};
    v8f acc[4][2];
#pragma unroll
    for (int mt = 0; mt < 4; ++mt)
#pragma unroll
        for (int nt = 0; nt < 2; ++nt) acc[mt][nt] = zero;

    const int row = tid >> 1;            // 0..127
    const int seg = (tid & 1) * 32;      // 0 or 32 halves
    const _Float16* ga = A  + (size_t)(mBlk + row) * K + seg;
    const _Float16* gb = Bw + (size_t)(nBlk + row) * K + seg;
    const int lofs = row * LDK + seg;

    const int nk = K / BK;               // 16

    // prologue: stage tile 0 into buffer 0 (8 async ops per wave)
#pragma unroll
    for (int j = 0; j < 4; ++j) {
        cp16_async_or_sync(ga + j * 8, &As[0][lofs] + j * 8);
        cp16_async_or_sync(gb + j * 8, &Bs[0][lofs] + j * 8);
    }

    for (int kt = 0; kt < nk; ++kt) {
        const int cur = kt & 1;
        __syncthreads();                 // all waves done reading buf[cur^1]
        if (kt + 1 < nk) {               // stage tile kt+1 into other buffer
            const int k1 = (kt + 1) * BK;
#pragma unroll
            for (int j = 0; j < 4; ++j) {
                cp16_async_or_sync(ga + k1 + j * 8, &As[cur ^ 1][lofs] + j * 8);
                cp16_async_or_sync(gb + k1 + j * 8, &Bs[cur ^ 1][lofs] + j * 8);
            }
            if (kt + 2 < nk) {           // gfx1250 global_prefetch_b8
                __builtin_prefetch(ga + k1 + BK, 0, 1);
                __builtin_prefetch(gb + k1 + BK, 0, 1);
            }
            WAIT_ASYNC(8);               // tile kt done; kt+1 still in flight
        } else {
            WAIT_ASYNC(0);
        }
        __syncthreads();                 // tile kt visible to all waves

        const _Float16* Asb = As[cur];
        const _Float16* Bsb = Bs[cur];
#pragma unroll
        for (int kc = 0; kc < BK; kc += 32) {
            v16h a[4];
#pragma unroll
            for (int mt = 0; mt < 4; ++mt)
                a[mt] = load_frag_a(Asb, LDK, wm * 64 + mt * 16, kc, lane);
#pragma unroll
            for (int nt = 0; nt < 2; ++nt) {
                v16h b = load_frag_b(Bsb, LDK, wn * 32 + nt * 16, kc, lane);
#pragma unroll
                for (int mt = 0; mt < 4; ++mt)
                    acc[mt][nt] = wmma_f16(a[mt], b, acc[mt][nt]);
            }
        }
    }

    // Epilogue: C element (r): row = r + 8*(lane>=16), col = lane%16
    const int hsel = (lane >> 4) & 1;
    const int colL = lane & 15;
#pragma unroll
    for (int mt = 0; mt < 4; ++mt) {
#pragma unroll
        for (int nt = 0; nt < 2; ++nt) {
#pragma unroll
            for (int r = 0; r < 8; ++r) {
                const int orow = mBlk + wm * 64 + mt * 16 + r + hsel * 8;
                const int ocol = nBlk + wn * 32 + nt * 16 + colL;
                const float v = acc[mt][nt][r] + bias[ocol];
                if (QKV) {
                    const int b = orow / T, t = orow % T;
                    const int h = ocol >> 6, d = ocol & 63;
                    ((_Float16*)OutP)
                        [(((size_t)(b * NHEAD + h)) * T + t) * HEAD_DIM + d] =
                        (_Float16)v;
                } else {
                    ((float*)OutP)[(size_t)orow * N + ocol] = v;
                }
            }
        }
    }
}

// ---------------------------------------------------------------------------
// Flash attention with stoichiometric bias.
// Grid: (T/64 q-tiles, B*NHEAD). Block: 128 threads = 4 wave32.
// ---------------------------------------------------------------------------
#define LDS_T 72   // 64 + 8 pad halves; 144B row stride, 16B aligned

__global__ __launch_bounds__(128) void attention_kernel(
    const _Float16* __restrict__ Qh, const _Float16* __restrict__ Kh,
    const _Float16* __restrict__ Vh, const float* __restrict__ frac,
    _Float16* __restrict__ AOut,
    const float* __restrict__ alpha_pos, const float* __restrict__ alpha_neg,
    const float* __restrict__ gamma, const float* __restrict__ delta,
    int T)
{
    __shared__ _Float16 Qs[64 * LDS_T];
    __shared__ _Float16 Ks[64 * LDS_T];
    __shared__ _Float16 Vt[64 * LDS_T];  // transposed: [d][key]
    __shared__ _Float16 Ps[64 * LDS_T];
    __shared__ float fq[64];
    __shared__ float fk[64];

    const int tid  = threadIdx.x;
    const int lane = tid & 31;
    const int wave = tid >> 5;                // 0..3
    const int bh   = blockIdx.y;
    const int b    = bh >> 4;                 // NHEAD = 16
    const int h    = bh & 15;
    const int q0   = blockIdx.x * 64;

    const float gs = gamma[0] * 0.125f;       // gamma * HEAD_DIM^-0.5
    const float dl = delta[0];
    const float ap = alpha_pos[h];
    const float an = alpha_neg[h];

    const size_t headOff = ((size_t)(b * NHEAD + h)) * T * HEAD_DIM;
    const int row = tid >> 1, seg = (tid & 1) * 32;

    {   // stage Q tile (async) + fq
        const _Float16* src = Qh + headOff + (size_t)(q0 + row) * HEAD_DIM + seg;
        _Float16* dst = Qs + row * LDS_T + seg;
#pragma unroll
        for (int j = 0; j < 4; ++j)
            cp16_async_or_sync(src + j * 8, dst + j * 8);
        if (tid < 64) fq[tid] = frac[b * T + q0 + tid];
        WAIT_ASYNC(0);
    }
    __syncthreads();

    const int hsel = (lane >> 4) & 1;
    const int colL = lane & 15;
    float fqr[8];
#pragma unroll
    for (int r = 0; r < 8; ++r) fqr[r] = fq[wave * 16 + r + hsel * 8];

    const v16h qa0 = load_frag_a(Qs, LDS_T, wave * 16, 0, lane);
    const v16h qa1 = load_frag_a(Qs, LDS_T, wave * 16, 32, lane);

    v8f zero = {};
    v8f o[4];
#pragma unroll
    for (int nt = 0; nt < 4; ++nt) o[nt] = zero;
    float mI[8], lI[8];
#pragma unroll
    for (int r = 0; r < 8; ++r) { mI[r] = -1.0e30f; lI[r] = 0.0f; }

    const int nKT = T / 64;
    for (int kt = 0; kt < nKT; ++kt) {
        __syncthreads();
        {   // stage K (async, row-major), V (sync, transposed), fk
            const size_t base = headOff + (size_t)(kt * 64 + row) * HEAD_DIM + seg;
            const _Float16* ks = Kh + base;
            _Float16* kd = Ks + row * LDS_T + seg;
#pragma unroll
            for (int j = 0; j < 4; ++j)
                cp16_async_or_sync(ks + j * 8, kd + j * 8);

            union { h8 h[4]; _Float16 e[32]; } uv;
            const h8* vs = (const h8*)(Vh + base);
            uv.h[0] = vs[0]; uv.h[1] = vs[1]; uv.h[2] = vs[2]; uv.h[3] = vs[3];
#pragma unroll
            for (int j = 0; j < 32; ++j)
                Vt[(seg + j) * LDS_T + row] = uv.e[j];
            if (tid < 64) fk[tid] = frac[b * T + kt * 64 + tid];
            if (kt + 1 < nKT) {   // prefetch next K/V tiles into caches
                const size_t nb = base + (size_t)64 * HEAD_DIM;
                __builtin_prefetch(Kh + nb, 0, 1);
                __builtin_prefetch(Vh + nb, 0, 1);
            }
            WAIT_ASYNC(0);
        }
        __syncthreads();

        // ---- S = gamma*scale * (Q K^T) + delta*stoich_bias ----
        float s[4][8];
#pragma unroll
        for (int nt = 0; nt < 4; ++nt) {
            v16h kb0 = load_frag_b(Ks, LDS_T, nt * 16, 0, lane);
            v16h kb1 = load_frag_b(Ks, LDS_T, nt * 16, 32, lane);
            v8f c = zero;
            c = wmma_f16(qa0, kb0, c);
            c = wmma_f16(qa1, kb1, c);
            const float fkv = fk[nt * 16 + colL];
#pragma unroll
            for (int r = 0; r < 8; ++r) {
                float D = fkv - fqr[r];
                D = fminf(0.2f, fmaxf(-0.2f, D));
                const float sb = ap * fmaxf(D, 0.0f) + an * fminf(D, 0.0f);
                s[nt][r] = gs * c[r] + dl * sb;
            }
        }

        // ---- online softmax (row reductions across 16-lane halves) ----
        float rm[8];
#pragma unroll
        for (int r = 0; r < 8; ++r)
            rm[r] = fmaxf(fmaxf(s[0][r], s[1][r]), fmaxf(s[2][r], s[3][r]));
#pragma unroll
        for (int off = 1; off < 16; off <<= 1)
#pragma unroll
            for (int r = 0; r < 8; ++r)
                rm[r] = fmaxf(rm[r], __shfl_xor(rm[r], off, 32));

        float al[8];
#pragma unroll
        for (int r = 0; r < 8; ++r) {
            const float mN = fmaxf(mI[r], rm[r]);
            al[r] = __expf(mI[r] - mN);
            mI[r] = mN;
        }
        float rs[8];
#pragma unroll
        for (int r = 0; r < 8; ++r) rs[r] = 0.0f;
#pragma unroll
        for (int nt = 0; nt < 4; ++nt)
#pragma unroll
            for (int r = 0; r < 8; ++r) {
                const float p = __expf(s[nt][r] - mI[r]);
                s[nt][r] = p;
                rs[r] += p;
            }
#pragma unroll
        for (int off = 1; off < 16; off <<= 1)
#pragma unroll
            for (int r = 0; r < 8; ++r)
                rs[r] += __shfl_xor(rs[r], off, 32);
#pragma unroll
        for (int r = 0; r < 8; ++r) lI[r] = lI[r] * al[r] + rs[r];
#pragma unroll
        for (int nt = 0; nt < 4; ++nt)
#pragma unroll
            for (int r = 0; r < 8; ++r) o[nt][r] *= al[r];

        // ---- write P (f16) to LDS, re-fragment as A ----
#pragma unroll
        for (int nt = 0; nt < 4; ++nt)
#pragma unroll
            for (int r = 0; r < 8; ++r)
                Ps[(wave * 16 + r + hsel * 8) * LDS_T + nt * 16 + colL] =
                    (_Float16)s[nt][r];
        __syncthreads();

        const v16h pa0 = load_frag_a(Ps, LDS_T, wave * 16, 0, lane);
        const v16h pa1 = load_frag_a(Ps, LDS_T, wave * 16, 32, lane);
#pragma unroll
        for (int nt = 0; nt < 4; ++nt) {
            v16h vb0 = load_frag_b(Vt, LDS_T, nt * 16, 0, lane);
            v16h vb1 = load_frag_b(Vt, LDS_T, nt * 16, 32, lane);
            o[nt] = wmma_f16(pa0, vb0, o[nt]);
            o[nt] = wmma_f16(pa1, vb1, o[nt]);
        }
    }

    // ---- normalize and store attn output [B][T][D_MODEL] as f16 ----
    float inv[8];
#pragma unroll
    for (int r = 0; r < 8; ++r) inv[r] = 1.0f / lI[r];
#pragma unroll
    for (int nt = 0; nt < 4; ++nt)
#pragma unroll
        for (int r = 0; r < 8; ++r) {
            const int rw = q0 + wave * 16 + r + hsel * 8;
            const int d  = nt * 16 + colL;
            AOut[((size_t)b * T + rw) * D_MODEL + h * HEAD_DIM + d] =
                (_Float16)(o[nt][r] * inv[r]);
        }
}

// ---------------------------------------------------------------------------
// Host-side launcher
// ---------------------------------------------------------------------------
extern "C" void kernel_launch(void* const* d_in, const int* in_sizes, int n_in,
                              void* d_out, int out_size, void* d_ws, size_t ws_size,
                              hipStream_t stream) {
    const float* query = (const float*)d_in[0];
    const float* key   = (const float*)d_in[1];
    const float* value = (const float*)d_in[2];
    const float* frac  = (const float*)d_in[3];
    const float* Wq    = (const float*)d_in[4];
    const float* bq    = (const float*)d_in[5];
    const float* Wk    = (const float*)d_in[6];
    const float* bk    = (const float*)d_in[7];
    const float* Wv    = (const float*)d_in[8];
    const float* bv    = (const float*)d_in[9];
    const float* Wo    = (const float*)d_in[10];
    const float* bo    = (const float*)d_in[11];
    const float* a_pos = (const float*)d_in[12];
    const float* a_neg = (const float*)d_in[13];
    const float* gamma = (const float*)d_in[14];
    const float* delta = (const float*)d_in[15];

    const int T  = 2048;                      // fixed by reference setup
    const int BT = in_sizes[3];               // B*T = 4096
    const int M  = BT;
    const int N  = D_MODEL, K = D_MODEL;
    const int B  = BT / T;

    const size_t elems = (size_t)BT * D_MODEL;   // 4M halves
    const size_t welm  = (size_t)D_MODEL * D_MODEL;
    _Float16* p  = (_Float16*)d_ws;
    _Float16* Xq = p; p += elems;
    _Float16* Xk = p; p += elems;
    _Float16* Xv = p; p += elems;
    _Float16* Qh = p; p += elems;
    _Float16* Kh = p; p += elems;
    _Float16* Vh = p; p += elems;
    _Float16* AO = p; p += elems;
    _Float16* Wqh = p; p += welm;
    _Float16* Wkh = p; p += welm;
    _Float16* Wvh = p; p += welm;
    _Float16* Woh = p; p += welm;

    // one-time f32 -> f16 conversion
    const int gcX = (int)(elems / 8 / 256);
    const int gcW = (int)(welm / 8 / 256);
    cvt_f32_f16_kernel<<<gcX, 256, 0, stream>>>(query, Xq, (int)elems);
    cvt_f32_f16_kernel<<<gcX, 256, 0, stream>>>(key,   Xk, (int)elems);
    cvt_f32_f16_kernel<<<gcX, 256, 0, stream>>>(value, Xv, (int)elems);
    cvt_f32_f16_kernel<<<gcW, 256, 0, stream>>>(Wq, Wqh, (int)welm);
    cvt_f32_f16_kernel<<<gcW, 256, 0, stream>>>(Wk, Wkh, (int)welm);
    cvt_f32_f16_kernel<<<gcW, 256, 0, stream>>>(Wv, Wvh, (int)welm);
    cvt_f32_f16_kernel<<<gcW, 256, 0, stream>>>(Wo, Woh, (int)welm);

    dim3 gridG(M / BM, N / BN);               // 32 x 8
    gemm_f16_kernel<true><<<gridG, 256, 0, stream>>>(Xq, Wqh, bq, Qh, M, N, K, T);
    gemm_f16_kernel<true><<<gridG, 256, 0, stream>>>(Xk, Wkh, bk, Kh, M, N, K, T);
    gemm_f16_kernel<true><<<gridG, 256, 0, stream>>>(Xv, Wvh, bv, Vh, M, N, K, T);

    dim3 gridA(T / 64, B * NHEAD);            // 32 x 32
    attention_kernel<<<gridA, 128, 0, stream>>>(Qh, Kh, Vh, frac, AO,
                                                a_pos, a_neg, gamma, delta, T);

    gemm_f16_kernel<false><<<gridG, 256, 0, stream>>>(AO, Woh, bo, d_out,
                                                      M, N, K, T);
}
